// SelfAttention_9715216024055
// MI455X (gfx1250) — matrix operand order repaired and checked by reading, not compiled, not run on hardware
//
#include <hip/hip_runtime.h>
#include <hip/hip_bf16.h>
#include <stdint.h>

// ---------------------------------------------------------------------------
// Self-attention (B=4, S=2048, E=1024, H=16, D=64) for gfx1250 / MI455X.
// bf16 WMMA (V_WMMA_F32_16X16X32_BF16) everywhere; flash-attention online
// softmax so the [B,H,S,S] energy tensor never touches memory; K tiles staged
// block-wide in LDS via GLOBAL_LOAD_ASYNC_TO_LDS_B128 (ASYNCcnt path).
// Workspace layout (bytes):
//   Xbf    @ 0         16 MiB   bf16 x            [8192,1024]
//   Wqkvbf @ 16 MiB     6 MiB   bf16 w_qkv        [3072,1024]
//   Woutbf @ 22 MiB     2 MiB   bf16 w_out        [1024,1024]
//   Qb     @ 24 MiB    16 MiB   bf16 Q            [B,H,S,D]
//   Kb     @ 40 MiB    16 MiB   bf16 K            [B,H,S,D]
//   Vb     @ 56 MiB    16 MiB   bf16 V            [B,H,S,D]
//   Abf    @ 72 MiB    16 MiB   bf16 attn output  [B,H,S,D] == faithful [B,S,E]
// Total ~88 MiB.
// ---------------------------------------------------------------------------

typedef __bf16 bf16;
typedef __attribute__((ext_vector_type(16))) __bf16        v16bf;
typedef __attribute__((ext_vector_type(8)))  float         v8f;
typedef __attribute__((ext_vector_type(4)))  unsigned int  u32x4;

union FragU {
  v16bf v;
  u32x4 q[2];
  bf16  e[16];
};

static constexpr int BATCH = 4;
static constexpr int SEQ   = 2048;
static constexpr int EMB   = 1024;
static constexpr int HEADS = 16;
static constexpr int HD    = 64;          // head dim
static constexpr int ROWS  = BATCH * SEQ; // 8192

__device__ __forceinline__ v8f wmma_bf16(v16bf a, v16bf b, v8f c) {
  // D = A(16x32) * B(32x16) + C, f32 accum
  return __builtin_amdgcn_wmma_f32_16x16x32_bf16(
      /*neg_a=*/false, a, /*neg_b=*/false, b,
      /*c_mod=*/(short)0, c, /*reuse_a=*/false, /*reuse_b=*/false);
}

// A-matrix fragment (16x32 MxK bf16) from a row-major source, leading dim lda.
// A layout (contiguous k, matching load_b_frag): lane L holds row M = L%16;
// lanes 0-15 hold K = k0+[0..15]; lanes 16-31 hold K = k0+[16..31].
// -> two aligned 16B loads per lane.
__device__ __forceinline__ v16bf load_a_frag(const bf16* A, int lda, int m0, int k0) {
  const int lane = threadIdx.x & 31;
  const int row  = m0 + (lane & 15);
  const int koff = k0 + ((lane >> 4) << 4); // +0 or +16
  FragU f;
  f.q[0] = *reinterpret_cast<const u32x4*>(A + (size_t)row * lda + koff);
  f.q[1] = *reinterpret_cast<const u32x4*>(A + (size_t)row * lda + koff + 8);
  return f.v;
}

// B-matrix fragment (32x16 KxN bf16) from a source where each B column is a
// contiguous run of K values (i.e. B^T row-major, leading dim ldb).
// Contiguous-k layout (matching load_a_frag): lane L holds column N = L%16; lanes 0-15 hold K = k0+[0..15],
// lanes 16-31 hold K = k0+[16..31]. -> one aligned 32B load per lane.
__device__ __forceinline__ v16bf load_b_frag(const bf16* Bt, int ldb, int n0, int k0) {
  const int lane = threadIdx.x & 31;
  const int row  = n0 + (lane & 15);
  const int koff = k0 + ((lane >> 4) << 4); // +0 or +16
  FragU f;
  const u32x4* p = reinterpret_cast<const u32x4*>(Bt + (size_t)row * ldb + koff);
  f.q[0] = p[0];
  f.q[1] = p[1];
  return f.v;
}

// Async global -> LDS copy of 16 bytes per lane (GLOBAL_LOAD_ASYNC_TO_LDS_B128,
// GV mode: VDST = LDS byte address, VADDR = 64-bit global address). Tracked by
// ASYNCcnt; pair with s_wait_asynccnt.
__device__ __forceinline__ void async_copy_b128(void* lds_ptr, const void* gptr) {
  const unsigned lds_addr = (unsigned)(uintptr_t)lds_ptr;   // low 32 bits = LDS offset
  const unsigned long long gaddr = (unsigned long long)(uintptr_t)gptr;
  asm volatile("global_load_async_to_lds_b128 %0, %1, off"
               :: "v"(lds_addr), "v"(gaddr) : "memory");
}

__device__ __forceinline__ void wait_asynccnt0() {
  asm volatile("s_wait_asynccnt 0x0" ::: "memory");
}

// ---------------------------------------------------------------------------
// Kernel 1: fp32 -> bf16 conversion (grid-stride)
// ---------------------------------------------------------------------------
__global__ __launch_bounds__(256) void f32_to_bf16_kernel(
    const float* __restrict__ src, bf16* __restrict__ dst, int n) {
  int i = blockIdx.x * blockDim.x + threadIdx.x;
  const int stride = gridDim.x * blockDim.x;
  for (; i < n; i += stride) dst[i] = (bf16)src[i];
}

// ---------------------------------------------------------------------------
// Kernel 2: QKV projection. C[8192,3072] = Xbf * Wqkv^T + b_qkv, scattered
// head-split into Q/K/V [B,H,S,D] bf16. One wave computes a 32x64 C tile
// (2 A frags x 4 B frags -> 8 WMMAs per k-step).
// ---------------------------------------------------------------------------
__global__ __launch_bounds__(256) void qkv_gemm_kernel(
    const bf16* __restrict__ X, const bf16* __restrict__ W,
    const float* __restrict__ bias,
    bf16* __restrict__ Q, bf16* __restrict__ K, bf16* __restrict__ V) {
  const int N = 3 * EMB;                                    // 3072
  const int ntiles = N / 64;                                // 48
  const int wave   = blockIdx.x * (blockDim.x >> 5) + (threadIdx.x >> 5);
  const int mt = wave / ntiles;
  const int nt = wave % ntiles;
  if (mt >= ROWS / 32) return;
  const int m0 = mt * 32, n0 = nt * 64;
  const int lane = threadIdx.x & 31;

  v8f acc[2][4] = {};
  for (int k0 = 0; k0 < EMB; k0 += 32) {
    if (k0 + 128 < EMB) {  // prefetch next A rows (global_prefetch_b8)
      __builtin_prefetch(X + (size_t)(m0 + (lane & 15)) * EMB + k0 + 128, 0, 1);
    }
    const v16bf a0 = load_a_frag(X, EMB, m0,      k0);
    const v16bf a1 = load_a_frag(X, EMB, m0 + 16, k0);
#pragma unroll
    for (int t = 0; t < 4; ++t) {
      const v16bf b = load_b_frag(W, EMB, n0 + 16 * t, k0);
      acc[0][t] = wmma_bf16(a0, b, acc[0][t]);
      acc[1][t] = wmma_bf16(a1, b, acc[1][t]);
    }
  }

  const int half8 = (lane >> 4) << 3;
  const int colb  = lane & 15;
#pragma unroll
  for (int t = 0; t < 4; ++t) {
    const int col   = n0 + 16 * t + colb;
    const float bv  = bias[col];
    const int three = col >> 10;          // 0=Q, 1=K, 2=V
    const int e     = col & 1023;
    const int h     = e >> 6;
    const int d     = e & 63;
    bf16* dst = (three == 0) ? Q : (three == 1) ? K : V;
#pragma unroll
    for (int i = 0; i < 2; ++i) {
#pragma unroll
      for (int r = 0; r < 8; ++r) {
        const int m = m0 + 16 * i + r + half8;
        const int b = m >> 11;            // /SEQ
        const int s = m & 2047;
        dst[(((size_t)(b * HEADS + h) * SEQ + s) * HD) + d] =
            (bf16)(acc[i][t][r] + bv);
      }
    }
  }
}

// ---------------------------------------------------------------------------
// Kernel 3: flash attention. One wave owns a 16-query tile; 4 waves per WG
// share (a) a K tile staged via async global->LDS copies, (b) an LDS-staged
// transposed V tile. Energies stay in registers.
// ---------------------------------------------------------------------------
__global__ __launch_bounds__(128) void flash_attn_kernel(
    const bf16* __restrict__ Q, const bf16* __restrict__ Km,
    const bf16* __restrict__ Vm, bf16* __restrict__ O) {
  __shared__ __align__(16) bf16 Klds[32][64];     // K tile, row-major (key, d)
  __shared__ __align__(16) bf16 Vt[64][32];       // Vt[d][k_local]: V^T tile
  __shared__ __align__(16) bf16 Plds[4][16][32];  // per-wave P staging

  const int lane = threadIdx.x & 31;
  const int wave = threadIdx.x >> 5;
  const int bh   = blockIdx.y;                     // b*H + h
  const int q0   = (blockIdx.x * 4 + wave) * 16;
  const bf16* Qb = Q  + (size_t)bh * SEQ * HD;
  const bf16* Kb = Km + (size_t)bh * SEQ * HD;
  const bf16* Vb = Vm + (size_t)bh * SEQ * HD;

  const v16bf qa0 = load_a_frag(Qb, HD, q0, 0);
  const v16bf qa1 = load_a_frag(Qb, HD, q0, 32);

  v8f o[4] = {};
  float mrow[8], lrow[8];
#pragma unroll
  for (int r = 0; r < 8; ++r) { mrow[r] = -1e30f; lrow[r] = 0.0f; }
  const float scale = 0.125f;  // 1/sqrt(64)

  for (int kt = 0; kt < SEQ; kt += 32) {
    __syncthreads();  // WAR: everyone done with previous Klds/Vt

    // Async-stage the shared K tile: 32x64 bf16 = 4 KiB = 256 x 16B chunks,
    // 2 chunks per thread, tracked on ASYNCcnt.
    {
#pragma unroll
      for (int c = 0; c < 2; ++c) {
        const int chunk = threadIdx.x + c * 128;   // 0..255
        const int krow  = chunk >> 3;              // 0..31
        const int dcol  = (chunk & 7) << 3;        // 0,8,...,56 (bf16 units)
        async_copy_b128(&Klds[krow][dcol],
                        Kb + (size_t)(kt + krow) * HD + dcol);
      }
    }

    // Cooperative V-tile transpose: thread tid loads 16 contiguous bf16 of one
    // V row (keys kt..kt+31) and scatters into Vt so B-columns become
    // K-contiguous for load_b_frag.
    {
      const int krow = threadIdx.x >> 2;           // 0..31
      const int dseg = (threadIdx.x & 3) << 4;     // 0,16,32,48
      FragU f;
      const u32x4* p = reinterpret_cast<const u32x4*>(Vb + (size_t)(kt + krow) * HD + dseg);
      f.q[0] = p[0];
      f.q[1] = p[1];
#pragma unroll
      for (int j = 0; j < 16; ++j) Vt[dseg + j][krow] = f.e[j];
    }

    wait_asynccnt0();   // this wave's async K copies landed in LDS
    __syncthreads();    // all waves' copies + V transpose visible

    // Energies for two 16-key tiles, B frags from the shared LDS K tile
    // (K rows are d-contiguous => direct B frags).
    v8f e0 = {}, e1 = {};
    {
      v16bf kb;
      kb = load_b_frag(&Klds[0][0], HD, 0,   0);  e0 = wmma_bf16(qa0, kb, e0);
      kb = load_b_frag(&Klds[0][0], HD, 0,  32);  e0 = wmma_bf16(qa1, kb, e0);
      kb = load_b_frag(&Klds[0][0], HD, 16,  0);  e1 = wmma_bf16(qa0, kb, e1);
      kb = load_b_frag(&Klds[0][0], HD, 16, 32);  e1 = wmma_bf16(qa1, kb, e1);
    }

    // Online softmax. C layout: lane holds rows M = r + (lane>=16?8:0), col
    // N = lane%16; row reductions are 16-lane (xor masks 1,2,4,8 stay in-half).
    const int half8 = (lane >> 4) << 3;
#pragma unroll
    for (int r = 0; r < 8; ++r) {
      float x0 = e0[r] * scale;
      float x1 = e1[r] * scale;
      float mloc = fmaxf(x0, x1);
      mloc = fmaxf(mloc, __shfl_xor(mloc, 1));
      mloc = fmaxf(mloc, __shfl_xor(mloc, 2));
      mloc = fmaxf(mloc, __shfl_xor(mloc, 4));
      mloc = fmaxf(mloc, __shfl_xor(mloc, 8));
      const float mnew  = fmaxf(mrow[r], mloc);
      const float alpha = __expf(mrow[r] - mnew);
      const float p0 = __expf(x0 - mnew);
      const float p1 = __expf(x1 - mnew);
      float ls = p0 + p1;
      ls += __shfl_xor(ls, 1);
      ls += __shfl_xor(ls, 2);
      ls += __shfl_xor(ls, 4);
      ls += __shfl_xor(ls, 8);
      lrow[r] = lrow[r] * alpha + ls;
      mrow[r] = mnew;
#pragma unroll
      for (int t = 0; t < 4; ++t) o[t][r] *= alpha;
      const int row = r + half8;
      Plds[wave][row][lane & 15]        = (bf16)p0;
      Plds[wave][row][16 + (lane & 15)] = (bf16)p1;
    }

    // O += P(16x32) * V(32x64): P via LDS round-trip (C-layout -> A-layout),
    // V columns from the transposed LDS tile.
    const v16bf pa = load_a_frag(&Plds[wave][0][0], 32, 0, 0);
#pragma unroll
    for (int t = 0; t < 4; ++t) {
      const v16bf vb = load_b_frag(&Vt[0][0], 32, t * 16, 0);
      o[t] = wmma_bf16(pa, vb, o[t]);
    }
  }

  // Epilogue: normalize and store [B,H,S,D] contiguous (== faithful reshape).
  const int half8 = (lane >> 4) << 3;
#pragma unroll
  for (int t = 0; t < 4; ++t) {
    const int d = t * 16 + (lane & 15);
#pragma unroll
    for (int r = 0; r < 8; ++r) {
      const int row = q0 + r + half8;
      const float val = o[t][r] / lrow[r];
      O[(size_t)bh * SEQ * HD + (size_t)row * HD + d] = (bf16)val;
    }
  }
}

// ---------------------------------------------------------------------------
// Kernel 4: output projection. out[8192,1024] = Abf * Wout^T + b_out (fp32 out)
// One wave computes a 32x64 C tile.
// ---------------------------------------------------------------------------
__global__ __launch_bounds__(256) void out_proj_kernel(
    const bf16* __restrict__ A, const bf16* __restrict__ W,
    const float* __restrict__ bias, float* __restrict__ out) {
  const int ntiles = EMB / 64;                              // 16
  const int wave   = blockIdx.x * (blockDim.x >> 5) + (threadIdx.x >> 5);
  const int mt = wave / ntiles;
  const int nt = wave % ntiles;
  if (mt >= ROWS / 32) return;
  const int m0 = mt * 32, n0 = nt * 64;
  const int lane = threadIdx.x & 31;

  v8f acc[2][4] = {};
  for (int k0 = 0; k0 < EMB; k0 += 32) {
    if (k0 + 128 < EMB) {
      __builtin_prefetch(A + (size_t)(m0 + (lane & 15)) * EMB + k0 + 128, 0, 1);
    }
    const v16bf a0 = load_a_frag(A, EMB, m0,      k0);
    const v16bf a1 = load_a_frag(A, EMB, m0 + 16, k0);
#pragma unroll
    for (int t = 0; t < 4; ++t) {
      const v16bf b = load_b_frag(W, EMB, n0 + 16 * t, k0);
      acc[0][t] = wmma_bf16(a0, b, acc[0][t]);
      acc[1][t] = wmma_bf16(a1, b, acc[1][t]);
    }
  }

  const int half8 = (lane >> 4) << 3;
#pragma unroll
  for (int t = 0; t < 4; ++t) {
    const int col  = n0 + 16 * t + (lane & 15);
    const float bv = bias[col];
#pragma unroll
    for (int i = 0; i < 2; ++i) {
#pragma unroll
      for (int r = 0; r < 8; ++r) {
        const int m = m0 + 16 * i + r + half8;
        out[(size_t)m * EMB + col] = acc[i][t][r] + bv;
      }
    }
  }
}

// ---------------------------------------------------------------------------
// Launch
// ---------------------------------------------------------------------------
extern "C" void kernel_launch(void* const* d_in, const int* in_sizes, int n_in,
                              void* d_out, int out_size, void* d_ws, size_t ws_size,
                              hipStream_t stream) {
  (void)in_sizes; (void)n_in; (void)out_size; (void)ws_size;

  const float* x     = (const float*)d_in[0];   // [4,2048,1024]
  const float* w_qkv = (const float*)d_in[1];   // [3072,1024]
  const float* b_qkv = (const float*)d_in[2];   // [3072]
  const float* w_out = (const float*)d_in[3];   // [1024,1024]
  const float* b_out = (const float*)d_in[4];   // [1024]
  float* out = (float*)d_out;                   // [4,2048,1024]

  char* ws = (char*)d_ws;
  bf16* Xbf    = (bf16*)(ws);
  bf16* Wqkvbf = (bf16*)(ws + (size_t)16777216);
  bf16* Woutbf = (bf16*)(ws + (size_t)23068672);
  bf16* Qb     = (bf16*)(ws + (size_t)25165824);
  bf16* Kb     = (bf16*)(ws + (size_t)41943040);
  bf16* Vb     = (bf16*)(ws + (size_t)58720256);
  bf16* Abf    = (bf16*)(ws + (size_t)75497472);

  // 1) fp32 -> bf16
  f32_to_bf16_kernel<<<2048, 256, 0, stream>>>(x, Xbf, ROWS * EMB);
  f32_to_bf16_kernel<<<1024, 256, 0, stream>>>(w_qkv, Wqkvbf, 3 * EMB * EMB);
  f32_to_bf16_kernel<<<512,  256, 0, stream>>>(w_out, Woutbf, EMB * EMB);

  // 2) QKV GEMM: (8192/32)*(3072/64) = 12288 wave-tiles, 8 waves/block
  qkv_gemm_kernel<<<1536, 256, 0, stream>>>(Xbf, Wqkvbf, b_qkv, Qb, Kb, Vb);

  // 3) flash attention: grid (S/64, B*H), 4 waves/block (one 16-query tile each)
  flash_attn_kernel<<<dim3(SEQ / 64, BATCH * HEADS), 128, 0, stream>>>(Qb, Kb, Vb, Abf);

  // 4) output projection: (8192/32)*(1024/64) = 4096 wave-tiles, 8 waves/block
  out_proj_kernel<<<512, 256, 0, stream>>>(Abf, Woutbf, b_out, out);
}
